// MambaCAN_2Direction_Fre_6803228196998
// MI455X (gfx1250) — matrix operand order repaired and checked by reading, not compiled, not run on hardware
//
#include <hip/hip_runtime.h>
#include <hip/hip_bf16.h>
#include <math.h>

typedef __attribute__((ext_vector_type(16))) _Float16 v16h;
typedef __attribute__((ext_vector_type(8)))  float    v8f;

#define LSEQ 100
#define DI   1024
#define NROW 51200   // 512 * 100

__device__ __forceinline__ float sigmoidf_(float x){ return 1.0f/(1.0f+__expf(-x)); }

// ---------------------------------------------------------------------------
// WMMA GEMM: C[M x N] = act(A[M x K] * W[N x K]^T + bias), f32 in/out,
// f16 multiply via v_wmma_f32_16x16x32_f16. Wave tile = 32(M) x NFRAG*16(N).
// Compile-time NFRAG/ACT/ACCUM -> fully unrolled, branch-free hot loop
// (no exec-mask save/restore, no accumulator copies).
// Requires M%32==0, K%32==0, N%(NFRAG*16)==0, all row strides even.
// ACT: 0=none 1=relu 2=softplus 3=silu. ACCUM: C += instead of C =.
// ---------------------------------------------------------------------------
template<int NFRAG, int ACT, bool ACCUM>
__global__ __launch_bounds__(128)
void gemm_wmma_f16(const float* __restrict__ A, int lda,
                   const float* __restrict__ W, int ldw,
                   float* __restrict__ C, int ldc,
                   int M, int N, int K,
                   const float* __restrict__ bias)
{
    constexpr int MFRAG = 2;
    int wid    = blockIdx.x * (blockDim.x >> 5) + (threadIdx.x >> 5);
    int tilesM = M >> 5;                       // 32 rows per wave
    int ngrp   = N / (NFRAG * 16);
    if (wid >= tilesM * ngrp) return;          // wave-uniform: EXEC stays all-ones
    int im = wid % tilesM;
    int ig = wid / tilesM;
    int m0 = im << 5, n0 = ig * (NFRAG * 16);

    int lane = threadIdx.x & 31;
    int l16  = lane & 15;
    int half = lane >> 4;

    v8f acc[MFRAG][NFRAG] = {};
    const float* Arow0 = A + (size_t)(m0 + l16) * lda;
    const float* Arow1 = Arow0 + (size_t)16 * lda;
    const float* Wlane = W + (size_t)(n0 + l16) * ldw;

    for (int k0 = 0; k0 < K; k0 += 32) {
        if (k0 + 64 <= K) __builtin_prefetch(Arow0 + k0 + 32, 0, 1);

        // ---- A fragments: 16x32 f16 each, per ISA 16-bit A layout ----
        v16h a[MFRAG];
        #pragma unroll
        for (int mf = 0; mf < MFRAG; ++mf) {
            const float* Ar = (mf == 0) ? Arow0 : Arow1;
            #pragma unroll
            for (int v = 0; v < 8; ++v) {
                int k = k0 + ((v >> 2) << 4) + (half << 3) + ((v & 3) << 1);
                float2 t = *(const float2*)(Ar + k);
                a[mf][2*v]   = (_Float16)t.x;
                a[mf][2*v+1] = (_Float16)t.y;
            }
        }
        // ---- B fragments (B = W^T, 32x16), each reused for both M frags ----
        #pragma unroll
        for (int f = 0; f < NFRAG; ++f) {
            const float* Wrow = Wlane + (size_t)(f * 16) * ldw + k0 + (half << 3);
            v16h bf;
            #pragma unroll
            for (int v = 0; v < 8; ++v) {
                float2 t = *(const float2*)(Wrow + ((v >> 2) << 4) + ((v & 3) << 1));
                bf[2*v]   = (_Float16)t.x;
                bf[2*v+1] = (_Float16)t.y;
            }
            #pragma unroll
            for (int mf = 0; mf < MFRAG; ++mf)
                acc[mf][f] = __builtin_amdgcn_wmma_f32_16x16x32_f16(
                                 false, a[mf], false, bf, (short)0,
                                 acc[mf][f], false, false);
        }
    }

    // ---- epilogue: C/D f32 layout (lane -> N, VGPR r -> M = r + 8*half) ----
    #pragma unroll
    for (int f = 0; f < NFRAG; ++f) {
        int n = n0 + f * 16 + l16;
        float bv = bias ? bias[n] : 0.f;
        #pragma unroll
        for (int mf = 0; mf < MFRAG; ++mf) {
            #pragma unroll
            for (int r = 0; r < 8; ++r) {
                int m = m0 + mf * 16 + r + (half << 3);
                float v = acc[mf][f][r] + bv;
                if (ACT == 1)      v = v > 0.f ? v : 0.f;
                else if (ACT == 2) v = v > 20.f ? v : logf(1.f + __expf(v));
                else if (ACT == 3) v = v * sigmoidf_(v);
                size_t idx = (size_t)m * ldc + n;
                if (ACCUM) C[idx] += v; else C[idx] = v;
            }
        }
    }
}

template<int NFRAG, int ACT, bool ACCUM>
static void launch_gemm(const float* A, int lda, const float* W, int ldw,
                        float* C, int ldc, int M, int N, int K,
                        const float* bias, hipStream_t stream)
{
    int waves  = (M >> 5) * (N / (NFRAG * 16));
    int blocks = (waves + 3) >> 2;
    gemm_wmma_f16<NFRAG, ACT, ACCUM><<<blocks, 128, 0, stream>>>(
        A, lda, W, ldw, C, ldc, M, N, K, bias);
}

// ---------------------------------------------------------------------------
// Frontend: id_embed = relu(LN2(relu(LN1(|x0|*fe_w + fe_b))))  -> comb[:, :256]
// ---------------------------------------------------------------------------
__global__ void fe_embed_kernel(const float* __restrict__ x,
                                const float* __restrict__ fe_w, const float* __restrict__ fe_b,
                                const float* __restrict__ g1,   const float* __restrict__ b1,
                                const float* __restrict__ g2,   const float* __restrict__ b2,
                                float* __restrict__ comb)
{
    __shared__ float s1[256], s2[256];
    int row = blockIdx.x;              // b*100 + l
    int b = row / LSEQ, l = row % LSEQ;
    int c = threadIdx.x;

    float idm = fabsf(x[(size_t)b * 2300 + l * 23]);
    float e = idm * fe_w[c] + fe_b[c];

    s1[c] = e; s2[c] = e * e; __syncthreads();
    for (int s = 128; s > 0; s >>= 1) {
        if (c < s) { s1[c] += s1[c+s]; s2[c] += s2[c+s]; }
        __syncthreads();
    }
    float mu = s1[0] / 256.f, var = s2[0] / 256.f - mu * mu;
    __syncthreads();

    float v = (e - mu) * rsqrtf(var + 1e-5f) * g1[c] + b1[c];
    v = v > 0.f ? v : 0.f;

    s1[c] = v; s2[c] = v * v; __syncthreads();
    for (int s = 128; s > 0; s >>= 1) {
        if (c < s) { s1[c] += s1[c+s]; s2[c] += s2[c+s]; }
        __syncthreads();
    }
    mu = s1[0] / 256.f; var = s2[0] / 256.f - mu * mu;

    float o = (v - mu) * rsqrtf(var + 1e-5f) * g2[c] + b2[c];
    o = o > 0.f ? o : 0.f;
    comb[(size_t)row * 512 + c] = o;
}

// conv1: data(B,22,100) -> f1(B,128,100), k=3 pad=1, relu
__global__ void conv1_kernel(const float* __restrict__ x,
                             const float* __restrict__ w, const float* __restrict__ bias,
                             float* __restrict__ f1)
{
    int b  = blockIdx.x;
    int co = blockIdx.y;
    int l  = threadIdx.x;
    if (l >= LSEQ) return;
    float acc = bias[co];
    const float* wr = w + (size_t)co * 22 * 3;
    #pragma unroll
    for (int j = 0; j < 3; ++j) {
        int t = l - 1 + j;
        if (t < 0 || t >= LSEQ) continue;
        const float* xr = x + (size_t)b * 2300 + t * 23 + 1;
        for (int ci = 0; ci < 22; ++ci)
            acc += wr[ci * 3 + j] * xr[ci];
    }
    f1[(size_t)b * 128 * LSEQ + (size_t)co * LSEQ + l] = acc > 0.f ? acc : 0.f;
}

// conv2: f1(B,128,100) -> comb[:, 256:512], k=3 pad=1, relu
__global__ void conv2_kernel(const float* __restrict__ f1,
                             const float* __restrict__ w, const float* __restrict__ bias,
                             float* __restrict__ comb)
{
    int b  = blockIdx.x;
    int co = blockIdx.y;
    int l  = threadIdx.x;
    if (l >= LSEQ) return;
    float acc = bias[co];
    const float* wr = w + (size_t)co * 128 * 3;
    const float* fb = f1 + (size_t)b * 128 * LSEQ;
    #pragma unroll
    for (int j = 0; j < 3; ++j) {
        int t = l - 1 + j;
        if (t < 0 || t >= LSEQ) continue;
        for (int ci = 0; ci < 128; ++ci)
            acc += wr[ci * 3 + j] * fb[ci * LSEQ + t];
    }
    comb[((size_t)b * LSEQ + l) * 512 + 256 + co] = acc > 0.f ? acc : 0.f;
}

// Depthwise causal conv + SiLU. rev=1 runs time backward (bwd mamba).
__global__ void dwconv_silu_kernel(const float* __restrict__ xin,
                                   const float* __restrict__ w,
                                   const float* __restrict__ bias,
                                   int Kc, int rev,
                                   float* __restrict__ xout)
{
    int idx = blockIdx.x * blockDim.x + threadIdx.x;   // b*DI + c
    int b = idx >> 10, c = idx & (DI - 1);
    const float* wr = w + (size_t)c * Kc;
    float bv = bias[c];
    size_t base = (size_t)b * LSEQ * DI + c;
    for (int l = 0; l < LSEQ; ++l) {
        float acc = bv;
        for (int j = 0; j < Kc; ++j) {
            int t = rev ? (l + (Kc - 1) - j) : (l - (Kc - 1) + j);
            if (t >= 0 && t < LSEQ)
                acc += wr[j] * xin[base + (size_t)t * DI];
        }
        xout[base + (size_t)l * DI] = acc * sigmoidf_(acc);
    }
}

// Selective scan: one thread per (b, channel). y may alias dt (read-before-write).
__global__ void scan_kernel(const float* dt, const float* __restrict__ xs,
                            const float* __restrict__ xdb, int ldx,
                            const float* __restrict__ Alog, int d_state,
                            int rev, float* y)
{
    int idx = blockIdx.x * blockDim.x + threadIdx.x;   // b*DI + c
    int b = idx >> 10, c = idx & (DI - 1);
    float h[16], A[16];
    for (int s = 0; s < d_state; ++s) { h[s] = 0.f; A[s] = -__expf(Alog[(size_t)c * d_state + s]); }
    for (int t = 0; t < LSEQ; ++t) {
        int l = rev ? (LSEQ - 1 - t) : t;
        size_t row = (size_t)b * LSEQ + l;
        float dtv = dt[row * DI + c];
        float xv  = xs[row * DI + c];
        const float* Bm = xdb + row * ldx + 32;
        const float* Cm = Bm + d_state;
        float dx = dtv * xv;
        float yv = 0.f;
        for (int s = 0; s < d_state; ++s) {
            float dA = __expf(dtv * A[s]);
            h[s] = dA * h[s] + dx * Bm[s];
            yv += h[s] * Cm[s];
        }
        y[row * DI + c] = yv;
    }
}

// y = (y + xs*D[c]) * zgate   (zgate already SiLU'd by the GEMM epilogue)
__global__ void gate_kernel(float* __restrict__ y, const float* __restrict__ xs,
                            const float* __restrict__ zg, const float* __restrict__ D,
                            size_t n)
{
    size_t i = (size_t)blockIdx.x * blockDim.x + threadIdx.x;
    if (i >= n) return;
    int c = (int)(i & (DI - 1));
    y[i] = (y[i] + xs[i] * D[c]) * zg[i];
}

// pooled[b] = LN(max_l acc[b,l,:], norm_g, norm_b)
__global__ void pool_ln_kernel(const float* __restrict__ acc,
                               const float* __restrict__ g, const float* __restrict__ b,
                               float* __restrict__ pooled)
{
    __shared__ float s1[512], s2[512];
    int bb = blockIdx.x, c = threadIdx.x;
    float m = -INFINITY;
    size_t base = (size_t)bb * LSEQ * 512 + c;
    for (int l = 0; l < LSEQ; ++l) m = fmaxf(m, acc[base + (size_t)l * 512]);
    s1[c] = m; s2[c] = m * m; __syncthreads();
    for (int s = 256; s > 0; s >>= 1) {
        if (c < s) { s1[c] += s1[c+s]; s2[c] += s2[c+s]; }
        __syncthreads();
    }
    float mu = s1[0] / 512.f, var = s2[0] / 512.f - mu * mu;
    pooled[(size_t)bb * 512 + c] = (m - mu) * rsqrtf(var + 1e-5f) * g[c] + b[c];
}

// fc2: (512x128) @ (4x128)^T + b
__global__ void fc2_kernel(const float* __restrict__ h, const float* __restrict__ w,
                           const float* __restrict__ b, float* __restrict__ out)
{
    int idx = blockIdx.x * blockDim.x + threadIdx.x;
    if (idx >= 512 * 4) return;
    int bb = idx >> 2, o = idx & 3;
    float acc = b[o];
    const float* hr = h + (size_t)bb * 128;
    const float* wr = w + (size_t)o * 128;
    for (int k = 0; k < 128; ++k) acc += hr[k] * wr[k];
    out[idx] = acc;
}

// ---------------------------------------------------------------------------
extern "C" void kernel_launch(void* const* d_in, const int* in_sizes, int n_in,
                              void* d_out, int out_size, void* d_ws, size_t ws_size,
                              hipStream_t stream)
{
    const float* x        = (const float*)d_in[0];
    const float* fe_w     = (const float*)d_in[1];
    const float* fe_b     = (const float*)d_in[2];
    const float* fe_ln_g  = (const float*)d_in[3];
    const float* fe_ln_b  = (const float*)d_in[4];
    const float* emb_ln_g = (const float*)d_in[5];
    const float* emb_ln_b = (const float*)d_in[6];
    const float* conv1_w  = (const float*)d_in[7];
    const float* conv1_b  = (const float*)d_in[8];
    const float* conv2_w  = (const float*)d_in[9];
    const float* conv2_b  = (const float*)d_in[10];
    const float* mf_in_w  = (const float*)d_in[11];
    const float* mf_cw    = (const float*)d_in[12];
    const float* mf_cb    = (const float*)d_in[13];
    const float* mf_xp    = (const float*)d_in[14];
    const float* mf_dtw   = (const float*)d_in[15];
    const float* mf_dtb   = (const float*)d_in[16];
    const float* mf_Alog  = (const float*)d_in[17];
    const float* mf_D     = (const float*)d_in[18];
    const float* mf_ow    = (const float*)d_in[19];
    const float* mb_in_w  = (const float*)d_in[20];
    const float* mb_cw    = (const float*)d_in[21];
    const float* mb_cb    = (const float*)d_in[22];
    const float* mb_xp    = (const float*)d_in[23];
    const float* mb_dtw   = (const float*)d_in[24];
    const float* mb_dtb   = (const float*)d_in[25];
    const float* mb_Alog  = (const float*)d_in[26];
    const float* mb_D     = (const float*)d_in[27];
    const float* mb_ow    = (const float*)d_in[28];
    const float* norm_g   = (const float*)d_in[29];
    const float* norm_b   = (const float*)d_in[30];
    const float* fc1_w    = (const float*)d_in[31];
    const float* fc1_b    = (const float*)d_in[32];
    const float* fc2_w    = (const float*)d_in[33];
    const float* fc2_b    = (const float*)d_in[34];
    (void)in_sizes; (void)n_in; (void)out_size; (void)ws_size;

    // Workspace carve-up (~880 MB, buffers aliased across phases):
    float* ws = (float*)d_ws;
    size_t off = 0;
    auto carve = [&](size_t n) { float* p = ws + off; off += n; return p; };
    float* comb   = carve((size_t)NROW * 512);   // frontend output (both dirs)
    float* accb   = carve((size_t)NROW * 512);   // fo then fo+bo
    float* bufX   = carve((size_t)NROW * DI);    // x_raw -> dt -> y (aliased)
    float* bufZ   = carve((size_t)NROW * DI);    // silu(z) gate
    float* bufS   = carve((size_t)NROW * DI);    // conv+silu output
    float* xdb    = carve((size_t)NROW * 64);    // x-proj output (dt-rank|B|C)
    float* f1     = carve((size_t)512 * 128 * LSEQ);
    float* pooled = carve((size_t)512 * 512);
    float* h1     = carve((size_t)512 * 128);

    // ---- frontend: comb = concat(id_embed, conv-features) ----
    fe_embed_kernel<<<NROW, 256, 0, stream>>>(x, fe_w, fe_b, fe_ln_g, fe_ln_b,
                                              emb_ln_g, emb_ln_b, comb);
    conv1_kernel<<<dim3(512, 128), 128, 0, stream>>>(x, conv1_w, conv1_b, f1);
    conv2_kernel<<<dim3(512, 256), 128, 0, stream>>>(f1, conv2_w, conv2_b, comb);

    // ---- one mamba direction ----
    auto run_mamba = [&](const float* in_w, const float* cw, const float* cb, int d_conv,
                         const float* xpw, int ldx, const float* dtw, const float* dtb,
                         const float* Alog, const float* Dp, const float* ow,
                         int d_state, int rev, int accum) {
        const int M = NROW;
        // in_proj split: x half (none), z half (fused SiLU)
        launch_gemm<4, 0, false>(comb, 512, in_w,                    512, bufX, DI,
                                 M, DI, 512, nullptr, stream);
        launch_gemm<4, 3, false>(comb, 512, in_w + (size_t)DI * 512, 512, bufZ, DI,
                                 M, DI, 512, nullptr, stream);
        // depthwise causal conv + SiLU (direction-aware)
        dwconv_silu_kernel<<<(512 * DI) / 256, 256, 0, stream>>>(bufX, cw, cb, d_conv, rev, bufS);
        // x-projection -> (dt-rank | B | C): N = 64 (fwd, NFRAG=4) or 48 (bwd, NFRAG=3)
        if (ldx == 64)
            launch_gemm<4, 0, false>(bufS, DI, xpw, DI, xdb, 64, M, 64, DI, nullptr, stream);
        else
            launch_gemm<3, 0, false>(bufS, DI, xpw, DI, xdb, 48, M, 48, DI, nullptr, stream);
        // dt = softplus(xdb[:, :32] @ dtw^T + dtb)  (overwrites bufX, now free)
        launch_gemm<4, 2, false>(xdb, ldx, dtw, 32, bufX, DI, M, DI, 32, dtb, stream);
        // selective scan (in-place dt -> y)
        scan_kernel<<<(512 * DI) / 256, 256, 0, stream>>>(bufX, bufS, xdb, ldx,
                                                          Alog, d_state, rev, bufX);
        // y = (y + xs*D) * silu(z)
        size_t nel = (size_t)M * DI;
        gate_kernel<<<(int)((nel + 255) / 256), 256, 0, stream>>>(bufX, bufS, bufZ, Dp, nel);
        // out_proj; accumulate backward pass on top of forward
        if (accum)
            launch_gemm<4, 0, true >(bufX, DI, ow, DI, accb, 512, M, 512, DI, nullptr, stream);
        else
            launch_gemm<4, 0, false>(bufX, DI, ow, DI, accb, 512, M, 512, DI, nullptr, stream);
    };

    run_mamba(mf_in_w, mf_cw, mf_cb, 4, mf_xp, 64, mf_dtw, mf_dtb, mf_Alog, mf_D, mf_ow, 16, 0, 0);
    run_mamba(mb_in_w, mb_cw, mb_cb, 2, mb_xp, 48, mb_dtw, mb_dtb, mb_Alog, mb_D, mb_ow, 8, 1, 1);

    // ---- head ----
    pool_ln_kernel<<<512, 512, 0, stream>>>(accb, norm_g, norm_b, pooled);
    launch_gemm<4, 1, false>(pooled, 512, fc1_w, 512, h1, 128, 512, 128, 512, fc1_b, stream);
    fc2_kernel<<<8, 256, 0, stream>>>(h1, fc2_w, fc2_b, (float*)d_out);
}
